// AttentionLayer_6794638262724
// MI455X (gfx1250) — compile-verified, matrix operand order checked by reading, not run on hardware
//
#include <hip/hip_runtime.h>

typedef __attribute__((ext_vector_type(16))) _Float16 v16h;
typedef __attribute__((ext_vector_type(8)))  _Float16 v8h;
typedef __attribute__((ext_vector_type(8)))  float    v8f;

#define T_DIM   512
#define D_DIM   256
#define A_DIM   50
#define A_PAD   64
#define XS      264            /* padded LDS row stride in halves: 528 B -> rows land on shifted banks */
#define NTHREADS 256
#define EPS_K   1e-7f

/* LDS layout (bytes) */
#define OFF_X    0
#define SZ_X     (T_DIM * XS * 2)            /* 270336 */
#define OFF_WT   (OFF_X + SZ_X)
#define SZ_WT    (A_PAD * XS * 2)            /* 33792  */
#define OFF_B    (OFF_WT + SZ_WT)
#define OFF_U    (OFF_B + A_PAD * 4)
#define OFF_S    (OFF_U + A_PAD * 4)
#define OFF_RED  (OFF_S + T_DIM * 4)
#define SMEM_BYTES (OFF_RED + NTHREADS * 4)  /* 307712 B < 320 KB */

__device__ __forceinline__ float fast_tanh(float x) {
    float e = __expf(2.0f * x);              /* v_exp_f32 */
    return (e - 1.0f) / (e + 1.0f);
}

__global__ __launch_bounds__(NTHREADS, 1)
void attn_pool_kernel(const float* __restrict__ x,
                      const float* __restrict__ W,
                      const float* __restrict__ bias,
                      const float* __restrict__ u,
                      float* __restrict__ out)
{
    extern __shared__ __align__(16) char smem[];
    _Float16* xs  = (_Float16*)(smem + OFF_X);
    _Float16* wt  = (_Float16*)(smem + OFF_WT);
    float*    bl  = (float*)(smem + OFF_B);
    float*    ul  = (float*)(smem + OFF_U);
    float*    sc  = (float*)(smem + OFF_S);
    float*    red = (float*)(smem + OFF_RED);

    const int tid  = threadIdx.x;
    const int b    = blockIdx.x;
    const int lane = tid & 31;           /* wave32 */
    const int wave = tid >> 5;

    /* ---- Phase 0: W^T (padded to 64 cols), bias, u into LDS ---- */
    for (int idx = tid; idx < A_PAD * D_DIM; idx += NTHREADS) {
        int n = idx & (A_PAD - 1);
        int k = idx >> 6;
        float w = (n < A_DIM) ? W[k * A_DIM + n] : 0.0f;
        wt[n * XS + k] = (_Float16)w;
    }
    if (tid < A_PAD) {
        bl[tid] = (tid < A_DIM) ? bias[tid] : 0.0f;
        ul[tid] = (tid < A_DIM) ? u[tid]    : 0.0f;
    }

    /* ---- Phase 1: x[b] fp32 -> f16 in LDS (the ONLY HBM pass over x) ---- */
    const float4* xg = (const float4*)(x + (size_t)b * T_DIM * D_DIM);
    for (int idx = tid; idx < T_DIM * (D_DIM / 4); idx += NTHREADS) {
        int t = idx >> 6;
        int c = idx & 63;
        float4 v = xg[t * (D_DIM / 4) + c];
        _Float16* dst = xs + t * XS + c * 4;
        dst[0] = (_Float16)v.x; dst[1] = (_Float16)v.y;
        dst[2] = (_Float16)v.z; dst[3] = (_Float16)v.w;
    }
    __syncthreads();

    /* ---- Phase 2: scores s[t] = tanh(x@W + b) . u  via v_wmma_f32_16x16x32_f16 ---- */
    /* A-fragment lane map (16-bit A 16x32): lane L -> row L%16, K-halves {8*(L/16)..+7, +16..+23} */
    const int mrow  = lane & 15;
    const int khalf = (lane >> 4) * 8;

    for (int mt = wave; mt < T_DIM / 16; mt += NTHREADS / 32) {
        const int t0 = mt * 16;
        v8f acc0 = {}, acc1 = {}, acc2 = {}, acc3 = {};

        for (int ks = 0; ks < D_DIM / 32; ++ks) {
            const _Float16* ap = xs + (t0 + mrow) * XS + ks * 32 + khalf;
            v16h a;
            *(v8h*)&a       = *(const v8h*)ap;         /* ds_load_b128 */
            *((v8h*)&a + 1) = *(const v8h*)(ap + 16);

            const _Float16* bp0 = wt + ( 0 + mrow) * XS + ks * 32 + khalf;
            const _Float16* bp1 = wt + (16 + mrow) * XS + ks * 32 + khalf;
            const _Float16* bp2 = wt + (32 + mrow) * XS + ks * 32 + khalf;
            const _Float16* bp3 = wt + (48 + mrow) * XS + ks * 32 + khalf;
            v16h b0, b1, b2, b3;
            *(v8h*)&b0 = *(const v8h*)bp0; *((v8h*)&b0 + 1) = *(const v8h*)(bp0 + 16);
            *(v8h*)&b1 = *(const v8h*)bp1; *((v8h*)&b1 + 1) = *(const v8h*)(bp1 + 16);
            *(v8h*)&b2 = *(const v8h*)bp2; *((v8h*)&b2 + 1) = *(const v8h*)(bp2 + 16);
            *(v8h*)&b3 = *(const v8h*)bp3; *((v8h*)&b3 + 1) = *(const v8h*)(bp3 + 16);

            acc0 = __builtin_amdgcn_wmma_f32_16x16x32_f16(false, a, false, b0, (short)0, acc0, false, false);
            acc1 = __builtin_amdgcn_wmma_f32_16x16x32_f16(false, a, false, b1, (short)0, acc1, false, false);
            acc2 = __builtin_amdgcn_wmma_f32_16x16x32_f16(false, a, false, b2, (short)0, acc2, false, false);
            acc3 = __builtin_amdgcn_wmma_f32_16x16x32_f16(false, a, false, b3, (short)0, acc3, false, false);
        }

        /* Epilogue on accumulators: C layout => VGPR r holds rows {r, r+8}, lane%16 = column */
        float part[8];
        #pragma unroll
        for (int r = 0; r < 8; ++r) part[r] = 0.0f;

        #pragma unroll
        for (int nt = 0; nt < 4; ++nt) {
            int   n  = nt * 16 + mrow;
            float bv = bl[n];
            float uv = ul[n];
            v8f c = (nt == 0) ? acc0 : (nt == 1) ? acc1 : (nt == 2) ? acc2 : acc3;
            #pragma unroll
            for (int r = 0; r < 8; ++r)
                part[r] += uv * fast_tanh(c[r] + bv);
        }

        /* reduce over the 16 columns held by lanes 0-15 (rows t0+r) and 16-31 (rows t0+8+r) */
        #pragma unroll
        for (int r = 0; r < 8; ++r) {
            float p = part[r];
            p += __shfl_xor(p, 8, 16);
            p += __shfl_xor(p, 4, 16);
            p += __shfl_xor(p, 2, 16);
            p += __shfl_xor(p, 1, 16);
            if (mrow == 0)
                sc[t0 + r + (lane >> 4) * 8] = p;
        }
    }
    __syncthreads();

    /* ---- Phase 3: exp-normalize over T (no max-subtract, matching reference) ---- */
    float e0 = __expf(sc[tid]);
    float e1 = __expf(sc[tid + NTHREADS]);
    red[tid] = e0 + e1;
    __syncthreads();
    #pragma unroll
    for (int s = NTHREADS / 2; s > 0; s >>= 1) {
        if (tid < s) red[tid] += red[tid + s];
        __syncthreads();
    }
    float inv = 1.0f / (red[0] + EPS_K);
    __syncthreads();
    sc[tid]            = e0 * inv;
    sc[tid + NTHREADS] = e1 * inv;
    __syncthreads();

    /* ---- Phase 4: weighted pool from LDS-resident x: out[b,d] = sum_t x[t,d]*ait[t] ---- */
    float acc = 0.0f;
    #pragma unroll 8
    for (int t = 0; t < T_DIM; ++t)
        acc += (float)xs[t * XS + tid] * sc[t];
    out[(size_t)b * D_DIM + tid] = acc;
}

extern "C" void kernel_launch(void* const* d_in, const int* in_sizes, int n_in,
                              void* d_out, int out_size, void* d_ws, size_t ws_size,
                              hipStream_t stream) {
    const float* x    = (const float*)d_in[0];   /* [256,512,256] */
    const float* W    = (const float*)d_in[1];   /* [256,50]      */
    const float* bias = (const float*)d_in[2];   /* [50]          */
    const float* u    = (const float*)d_in[3];   /* [50,1]        */
    float* out        = (float*)d_out;           /* [256,256]     */

    hipFuncSetAttribute((const void*)attn_pool_kernel,
                        hipFuncAttributeMaxDynamicSharedMemorySize, SMEM_BYTES);
    attn_pool_kernel<<<256, NTHREADS, SMEM_BYTES, stream>>>(x, W, bias, u, out);
}
